// KeywordsGuidedDecoder_21449066676719
// MI455X (gfx1250) — compile-verified, matrix operand order checked by reading, not run on hardware
//
#include <hip/hip_runtime.h>
#include <hip/hip_bf16.h>

typedef __attribute__((ext_vector_type(16))) __bf16 v16bf;
typedef __attribute__((ext_vector_type(8)))  __bf16 v8bf;
typedef __attribute__((ext_vector_type(8)))  float  v8f;
typedef __attribute__((ext_vector_type(4)))  float  v4f;
typedef __attribute__((ext_vector_type(4)))  unsigned int v4u;
typedef __attribute__((ext_vector_type(8)))  int v8i;
typedef __attribute__((ext_vector_type(4)))  int v4i;

#define D_MODEL 1024
#define D_HEAD  64

// ---------------------------------------------------------------------------
// WMMA fragment loaders.  Per CDNA5 ISA 7.12.2 (16-bit A/B 16x32 fragment):
// lane<16:  row = lane,    elems 0..7 = K 0..7,  elems 8..15 = K 16..23
// lane>=16: row = lane-16, elems 0..7 = K 8..15, elems 8..15 = K 24..31
// ---------------------------------------------------------------------------
__device__ __forceinline__ v16bf ld_frag_bf(const __bf16* base, int ldk) {
  const int lane = threadIdx.x & 31;
  const int r    = lane & 15;
  const int ko   = (lane & 16) ? 8 : 0;
  const __bf16* p = base + r * ldk + ko;
  v8bf lo = *reinterpret_cast<const v8bf*>(p);
  v8bf hi = *reinterpret_cast<const v8bf*>(p + 16);
  v16bf f;
#pragma unroll
  for (int i = 0; i < 8; ++i) { f[i] = lo[i]; f[i + 8] = hi[i]; }
  return f;
}

// Same fragment pattern but sourced from an f32 LDS tile, converting to bf16.
__device__ __forceinline__ v16bf ld_frag_f32(const float* base, int ldk) {
  const int lane = threadIdx.x & 31;
  const int r    = lane & 15;
  const int ko   = (lane & 16) ? 8 : 0;
  const float* p = base + r * ldk + ko;
  v4f a0 = *reinterpret_cast<const v4f*>(p);
  v4f a1 = *reinterpret_cast<const v4f*>(p + 4);
  v4f b0 = *reinterpret_cast<const v4f*>(p + 16);
  v4f b1 = *reinterpret_cast<const v4f*>(p + 20);
  v16bf f;
#pragma unroll
  for (int i = 0; i < 4; ++i) {
    f[i]      = (__bf16)a0[i];
    f[4 + i]  = (__bf16)a1[i];
    f[8 + i]  = (__bf16)b0[i];
    f[12 + i] = (__bf16)b1[i];
  }
  return f;
}

// ---------------------------------------------------------------------------
// Tensor Data Mover: DMA one 2-D f32 tile (tile_w x tile_h, row stride ld,
// all in elements) from global memory into LDS at byte offset lds_off.
// D# packing per CDNA5 ISA 8.3-8.6.  tensor_dim1 = 2^20 so rows are never OOB;
// tensor_dim0 = row length for the X bound.  Groups 2/3 unused (2-D tensor).
// This toolchain exposes the 6-arg builtin (g0, g1, g2, g3, g4, cpol).
// ---------------------------------------------------------------------------
__device__ __forceinline__ void tdm_load_2d_f32(unsigned lds_off, const float* gptr,
                                                int tile_w, int tile_h,
                                                int ld, int tensor_w) {
  unsigned long long ga = (unsigned long long)(uintptr_t)gptr;
  v4u g0;
  g0[0] = 1u;                                            // count=1, user mode
  g0[1] = lds_off;                                       // lds_addr (bytes)
  g0[2] = (unsigned)(ga & 0xFFFFFFFFu);                  // global_addr[31:0]
  g0[3] = (unsigned)((ga >> 32) & 0x1FFFFFFu) | (2u << 30); // addr[56:32] | type=2
  v8i g1;
  g1[0] = (int)(2u << 16);                               // data_size = 4 bytes
  g1[1] = (int)(((unsigned)tensor_w & 0xFFFFu) << 16);   // tensor_dim0 lo16
  g1[2] = (int)(((unsigned)tensor_w >> 16) & 0xFFFFu);   // tensor_dim0 hi16, dim1 lo16=0
  g1[3] = (int)(0x10u | ((unsigned)tile_w << 16));       // dim1 hi16 (2^20), tile_dim0
  g1[4] = (int)((unsigned)tile_h & 0xFFFFu);             // tile_dim1, tile_dim2=0
  g1[5] = (int)(unsigned)ld;                             // tensor_dim0_stride lo32
  g1[6] = 0;                                             // stride0 hi, stride1 lo
  g1[7] = 0;
  v4i gz4 = {0, 0, 0, 0};
  v8i gz8 = {0, 0, 0, 0, 0, 0, 0, 0};
  __builtin_amdgcn_tensor_load_to_lds(g0, g1, gz4, gz4, gz8, 0);
}

// ---------------------------------------------------------------------------
// C = A(f32,[M,K]) @ B(f32,[K,N]) (+bias)(+relu) -> f32 [M,N]
// Tiles staged by the Tensor Data Mover, bf16 WMMA, f32 accumulate.
// M,N multiples of 64; K multiple of 32.
// ---------------------------------------------------------------------------
__global__ __launch_bounds__(128)
void gemm_bf16_wmma(const float* __restrict__ A, const float* __restrict__ B,
                    const float* __restrict__ bias, float* __restrict__ C,
                    int M, int N, int K, int relu) {
  __shared__ __align__(16) float  AsF[64 * 32];   // [m][k]  f32 (TDM dest)
  __shared__ __align__(16) float  BsF[32 * 64];   // [k][n]  f32 (TDM dest)
  __shared__ __align__(16) __bf16 Bs[64 * 32];    // [n][k]  bf16 (transposed)
  const int tid  = threadIdx.x;
  const int wave = tid >> 5;
  const int lane = tid & 31;
  const int n0 = blockIdx.x * 64;
  const int m0 = blockIdx.y * 64;

  const unsigned asf_off = (unsigned)(uintptr_t)(void*)AsF;
  const unsigned bsf_off = (unsigned)(uintptr_t)(void*)BsF;

  v8f acc[4] = {};

  for (int k0 = 0; k0 < K; k0 += 32) {
    __syncthreads();                       // previous-iteration LDS reads done
    if (wave == 0) {
      tdm_load_2d_f32(asf_off, A + (size_t)m0 * K + k0, 32, 64, K, K);
      tdm_load_2d_f32(bsf_off, B + (size_t)k0 * N + n0, 64, 32, N, N);
      __builtin_amdgcn_s_wait_tensorcnt(0);
    }
    __syncthreads();                       // tiles visible to all waves

    // Transpose + convert B: BsF[k][n] f32 -> Bs[n][k] bf16 (16 elems/thread)
    {
      const int k  = tid >> 2;             // 0..31
      const int nq = tid & 3;              // 16-wide column chunk
#pragma unroll
      for (int i = 0; i < 16; ++i) {
        int n = nq * 16 + i;
        Bs[n * 32 + k] = (__bf16)BsF[k * 64 + n];
      }
    }
    __syncthreads();

    v16bf a = ld_frag_f32(&AsF[wave * 16 * 32], 32);
#pragma unroll
    for (int nt = 0; nt < 4; ++nt) {
      v16bf b = ld_frag_bf(&Bs[nt * 16 * 32], 32);
      acc[nt] = __builtin_amdgcn_wmma_f32_16x16x32_bf16(
          false, a, false, b, (short)0, acc[nt], false, false);
    }
  }

  const int half8 = (lane & 16) ? 8 : 0;
  const int cn    = lane & 15;
#pragma unroll
  for (int nt = 0; nt < 4; ++nt) {
#pragma unroll
    for (int e = 0; e < 8; ++e) {
      int row = m0 + wave * 16 + e + half8;
      int col = n0 + nt * 16 + cn;
      float v = acc[nt][e];
      if (bias) v += bias[col];
      if (relu) v = fmaxf(v, 0.0f);
      C[(size_t)row * N + col] = v;
    }
  }
}

// ---------------------------------------------------------------------------
// Fused flash attention, one wave per (b, h, 16-row q tile).
// Q,K,V,O are f32 [B, T*, D_MODEL]; head h uses columns [h*64, h*64+64).
// causal=1: key <= query (dec_valid_lens = q+1).  causal=0: key < vlen[b].
// ---------------------------------------------------------------------------
__global__ __launch_bounds__(32)
void attn_wmma(const float* __restrict__ Q, const float* __restrict__ Kt,
               const float* __restrict__ V, const int* __restrict__ vlen,
               float* __restrict__ O, int Tq, int Tk, int causal) {
  __shared__ __align__(16) __bf16 Qs[16 * 64];   // [q][d]
  __shared__ __align__(16) __bf16 Ks[32 * 64];   // [key][d] (== B-frag layout, QK^T)
  __shared__ __align__(16) __bf16 Vt[64 * 32];   // [d][key] (B-frag layout, PV)
  __shared__ __align__(16) __bf16 Ps[16 * 32];   // [q][key] P relayout buffer

  const int lane = threadIdx.x & 31;
  const int qb   = blockIdx.x * 16;
  const int h    = blockIdx.y;
  const int b    = blockIdx.z;
  const int half8 = (lane & 16) ? 8 : 0;
  const int cn    = lane & 15;

  const size_t qoff = ((size_t)b * Tq + qb) * D_MODEL + (size_t)h * D_HEAD;
#pragma unroll
  for (int i = 0; i < 32; ++i) {
    int idx = i * 32 + lane;
    int r = idx >> 6, c = idx & 63;
    Qs[r * 64 + c] = (__bf16)Q[qoff + (size_t)r * D_MODEL + c];
  }
  v16bf aq0 = ld_frag_bf(&Qs[0], 64);
  v16bf aq1 = ld_frag_bf(&Qs[32], 64);

  float m[8], l[8];
#pragma unroll
  for (int e = 0; e < 8; ++e) { m[e] = -3.0e38f; l[e] = 0.0f; }
  v8f o0 = {}, o1 = {}, o2 = {}, o3 = {};

  const int vl   = causal ? Tk : vlen[b];
  const int nblk = causal ? ((qb + 16 + 31) >> 5) : ((vl + 31) >> 5);

  for (int j = 0; j < nblk; ++j) {
    const int kb = j * 32;
    for (int i = 0; i < 64; ++i) {
      int idx = i * 32 + lane;
      int kk = idx >> 6, c = idx & 63;
      size_t g = ((size_t)b * Tk + kb + kk) * D_MODEL + (size_t)h * D_HEAD + c;
      float kv = Kt[g], vv = V[g];
      Ks[kk * 64 + c] = (__bf16)kv;
      Vt[c * 32 + kk] = (__bf16)vv;
    }

    // S = Q @ K^T for 32 keys (two 16-key column tiles)
    v8f s0 = {}, s1 = {};
    {
      v16bf b0a = ld_frag_bf(&Ks[0], 64);
      v16bf b0b = ld_frag_bf(&Ks[32], 64);
      v16bf b1a = ld_frag_bf(&Ks[16 * 64], 64);
      v16bf b1b = ld_frag_bf(&Ks[16 * 64 + 32], 64);
      s0 = __builtin_amdgcn_wmma_f32_16x16x32_bf16(false, aq0, false, b0a, (short)0, s0, false, false);
      s0 = __builtin_amdgcn_wmma_f32_16x16x32_bf16(false, aq1, false, b0b, (short)0, s0, false, false);
      s1 = __builtin_amdgcn_wmma_f32_16x16x32_bf16(false, aq0, false, b1a, (short)0, s1, false, false);
      s1 = __builtin_amdgcn_wmma_f32_16x16x32_bf16(false, aq1, false, b1b, (short)0, s1, false, false);
    }

    float p0[8], p1[8];
#pragma unroll
    for (int e = 0; e < 8; ++e) {
      float s0e = s0[e] * 0.125f;   // 1/sqrt(64)
      float s1e = s1[e] * 0.125f;
      int row = qb + e + half8;
      int c0 = kb + cn, c1 = kb + 16 + cn;
      bool ok0 = causal ? (c0 <= row) : (c0 < vl);
      bool ok1 = causal ? (c1 <= row) : (c1 < vl);
      if (!ok0) s0e = -1.0e6f;
      if (!ok1) s1e = -1.0e6f;
      float rm = fmaxf(s0e, s1e);
#pragma unroll
      for (int msk = 1; msk < 16; msk <<= 1) rm = fmaxf(rm, __shfl_xor(rm, msk, 32));
      float mn    = fmaxf(m[e], rm);
      float alpha = __expf(m[e] - mn);
      float e0 = __expf(s0e - mn);
      float e1 = __expf(s1e - mn);
      float rs = e0 + e1;
#pragma unroll
      for (int msk = 1; msk < 16; msk <<= 1) rs += __shfl_xor(rs, msk, 32);
      l[e] = l[e] * alpha + rs;
      m[e] = mn;
      o0[e] *= alpha; o1[e] *= alpha; o2[e] *= alpha; o3[e] *= alpha;
      p0[e] = e0; p1[e] = e1;
    }

    // Relayout P (C layout) -> A-fragment layout through LDS
#pragma unroll
    for (int e = 0; e < 8; ++e) {
      Ps[(e + half8) * 32 + cn]      = (__bf16)p0[e];
      Ps[(e + half8) * 32 + 16 + cn] = (__bf16)p1[e];
    }
    v16bf ap = ld_frag_bf(&Ps[0], 32);
    o0 = __builtin_amdgcn_wmma_f32_16x16x32_bf16(false, ap, false, ld_frag_bf(&Vt[0],       32), (short)0, o0, false, false);
    o1 = __builtin_amdgcn_wmma_f32_16x16x32_bf16(false, ap, false, ld_frag_bf(&Vt[16 * 32], 32), (short)0, o1, false, false);
    o2 = __builtin_amdgcn_wmma_f32_16x16x32_bf16(false, ap, false, ld_frag_bf(&Vt[32 * 32], 32), (short)0, o2, false, false);
    o3 = __builtin_amdgcn_wmma_f32_16x16x32_bf16(false, ap, false, ld_frag_bf(&Vt[48 * 32], 32), (short)0, o3, false, false);
  }

#pragma unroll
  for (int e = 0; e < 8; ++e) {
    float inv = 1.0f / l[e];
    size_t ro = ((size_t)b * Tq + qb + e + half8) * D_MODEL + (size_t)h * D_HEAD;
    O[ro + cn]      = o0[e] * inv;
    O[ro + 16 + cn] = o1[e] * inv;
    O[ro + 32 + cn] = o2[e] * inv;
    O[ro + 48 + cn] = o3[e] * inv;
  }
}

// out = LayerNorm(a + r) * g + bt, one block per row of D_MODEL
__global__ __launch_bounds__(256)
void add_ln(const float* __restrict__ a, const float* __restrict__ r,
            const float* __restrict__ g, const float* __restrict__ bt,
            float* __restrict__ out) {
  __shared__ float red[256];
  const int tid = threadIdx.x;
  const size_t row = blockIdx.x;
  float v[4];
  float s = 0.0f;
#pragma unroll
  for (int i = 0; i < 4; ++i) {
    int c = tid + i * 256;
    v[i] = a[row * D_MODEL + c] + r[row * D_MODEL + c];
    s += v[i];
  }
  red[tid] = s; __syncthreads();
  for (int off = 128; off > 0; off >>= 1) { if (tid < off) red[tid] += red[tid + off]; __syncthreads(); }
  float mean = red[0] * (1.0f / D_MODEL);
  __syncthreads();
  float sq = 0.0f;
#pragma unroll
  for (int i = 0; i < 4; ++i) { float d = v[i] - mean; sq += d * d; }
  red[tid] = sq; __syncthreads();
  for (int off = 128; off > 0; off >>= 1) { if (tid < off) red[tid] += red[tid + off]; __syncthreads(); }
  float rstd = rsqrtf(red[0] * (1.0f / D_MODEL) + 1e-5f);
#pragma unroll
  for (int i = 0; i < 4; ++i) {
    int c = tid + i * 256;
    out[row * D_MODEL + c] = (v[i] - mean) * rstd * g[c] + bt[c];
  }
}

// gate = sigmoid(concat(code,kw) @ gw + gb); out = gate*code + (1-gate)*kw
__global__ __launch_bounds__(256)
void gate_mix(const float* __restrict__ code, const float* __restrict__ kw,
              const float* __restrict__ gw, const float* __restrict__ gb,
              float* __restrict__ out) {
  __shared__ float red[256];
  const int tid = threadIdx.x;
  const size_t row = blockIdx.x;
  float s = 0.0f;
#pragma unroll
  for (int i = 0; i < 4; ++i) {
    int c = tid + i * 256;
    s += code[row * D_MODEL + c] * gw[c] + kw[row * D_MODEL + c] * gw[D_MODEL + c];
  }
  red[tid] = s; __syncthreads();
  for (int off = 128; off > 0; off >>= 1) { if (tid < off) red[tid] += red[tid + off]; __syncthreads(); }
  float gsc = 1.0f / (1.0f + __expf(-(red[0] + gb[0])));
#pragma unroll
  for (int i = 0; i < 4; ++i) {
    int c = tid + i * 256;
    float cv = code[row * D_MODEL + c];
    float kv = kw[row * D_MODEL + c];
    out[row * D_MODEL + c] = gsc * cv + (1.0f - gsc) * kv;
  }
}

extern "C" void kernel_launch(void* const* d_in, const int* in_sizes, int n_in,
                              void* d_out, int out_size, void* d_ws, size_t ws_size,
                              hipStream_t stream) {
  const int B = 8, T = 512, SC = 512, ST = 256, SK = 64, D = 1024, DFF = 4096;
  const float* x       = (const float*)d_in[0];
  const float* src     = (const float*)d_in[1];
  const int*   src_len = (const int*)  d_in[2];
  const float* tpl     = (const float*)d_in[3];
  const int*   tpl_len = (const int*)  d_in[4];
  const float* kwe     = (const float*)d_in[5];
  const int*   kw_len  = (const int*)  d_in[6];
  const float* sa_w    = (const float*)d_in[7];
  const float* cc_w    = (const float*)d_in[8];
  const float* ck_w    = (const float*)d_in[9];
  const float* ct_w    = (const float*)d_in[10];
  const float* gate_w  = (const float*)d_in[11];
  const float* gate_b  = (const float*)d_in[12];
  const float* w1      = (const float*)d_in[13];
  const float* b1      = (const float*)d_in[14];
  const float* w2      = (const float*)d_in[15];
  const float* b2      = (const float*)d_in[16];
  const float* ln_g    = (const float*)d_in[17];
  const float* ln_b    = (const float*)d_in[18];

  const size_t DD   = (size_t)D * D;
  const size_t NTOK = (size_t)B * T * D;   // 4M floats

  float* ws = (float*)d_ws;
  size_t off = 0;
  auto take = [&](size_t n) { float* p = ws + off; off += n; return p; };
  float* Qb = take(NTOK); float* Kb = take(NTOK); float* Vb = take(NTOK);
  float* Ab = take(NTOK); float* T1 = take(NTOK); float* T2 = take(NTOK);
  float* Yb = take(NTOK); float* Cb = take(NTOK); float* Zb = take(NTOK);
  float* Wb = take(NTOK); float* Hb = take((size_t)B * T * DFF);

  auto gemm = [&](const float* A, const float* Bm, const float* bias, float* C,
                  int M, int N, int K, int relu) {
    dim3 grid(N / 64, M / 64);
    gemm_bf16_wmma<<<grid, dim3(128), 0, stream>>>(A, Bm, bias, C, M, N, K, relu);
  };
  auto attn = [&](const float* Qp, const float* Kp, const float* Vp, const int* vl,
                  float* Op, int Tk, int causal) {
    dim3 grid(T / 16, 16, B);
    attn_wmma<<<grid, dim3(32), 0, stream>>>(Qp, Kp, Vp, vl, Op, T, Tk, causal);
  };
  auto lnk = [&](const float* a, const float* r, int li, float* o) {
    add_ln<<<dim3(B * T), dim3(256), 0, stream>>>(a, r, ln_g + (size_t)li * D,
                                                  ln_b + (size_t)li * D, o);
  };

  // ---- causal self-attention ----
  gemm(x, sa_w + 0 * DD, nullptr, Qb, B * T, D, D, 0);
  gemm(x, sa_w + 1 * DD, nullptr, Kb, B * T, D, D, 0);
  gemm(x, sa_w + 2 * DD, nullptr, Vb, B * T, D, D, 0);
  attn(Qb, Kb, Vb, nullptr, Ab, T, 1);
  gemm(Ab, sa_w + 3 * DD, nullptr, T1, B * T, D, D, 0);
  lnk(x, T1, 0, Yb);                              // y

  // ---- cross-attention: source code ----
  gemm(Yb,  cc_w + 0 * DD, nullptr, Qb, B * T,  D, D, 0);
  gemm(src, cc_w + 1 * DD, nullptr, Kb, B * SC, D, D, 0);
  gemm(src, cc_w + 2 * DD, nullptr, Vb, B * SC, D, D, 0);
  attn(Qb, Kb, Vb, src_len, Ab, SC, 0);
  gemm(Ab, cc_w + 3 * DD, nullptr, Cb, B * T, D, D, 0);   // y2_code

  // ---- cross-attention: keywords ----
  gemm(Yb,  ck_w + 0 * DD, nullptr, Qb, B * T,  D, D, 0);
  gemm(kwe, ck_w + 1 * DD, nullptr, Kb, B * SK, D, D, 0);
  gemm(kwe, ck_w + 2 * DD, nullptr, Vb, B * SK, D, D, 0);
  attn(Qb, Kb, Vb, kw_len, Ab, SK, 0);
  gemm(Ab, ck_w + 3 * DD, nullptr, T1, B * T, D, D, 0);   // y2_kw

  gate_mix<<<dim3(B * T), dim3(256), 0, stream>>>(Cb, T1, gate_w, gate_b, T2);
  lnk(Yb, T2, 1, Zb);                             // z

  // ---- cross-attention: template ----
  gemm(Zb,  ct_w + 0 * DD, nullptr, Qb, B * T,  D, D, 0);
  gemm(tpl, ct_w + 1 * DD, nullptr, Kb, B * ST, D, D, 0);
  gemm(tpl, ct_w + 2 * DD, nullptr, Vb, B * ST, D, D, 0);
  attn(Qb, Kb, Vb, tpl_len, Ab, ST, 0);
  gemm(Ab, ct_w + 3 * DD, nullptr, T1, B * T, D, D, 0);   // z2
  lnk(Zb, T1, 2, Wb);                             // w

  // ---- FFN ----
  gemm(Wb, w1, b1, Hb, B * T, DFF, D, 1);         // relu(w@w1+b1)
  gemm(Hb, w2, b2, T1, B * T, D, DFF, 0);
  lnk(Wb, T1, 3, (float*)d_out);                  // final LN

  (void)in_sizes; (void)n_in; (void)out_size; (void)ws_size;
}
